// HGT_15659450761478
// MI455X (gfx1250) — matrix operand order repaired and checked by reading, not compiled, not run on hardware
//
#include <hip/hip_runtime.h>
#include <hip/hip_bf16.h>
#include <math.h>

// ---------------- problem constants ----------------
#define NI_  40000
#define NV_  20000
#define NC_  8000
#define NB_  4000
#define NTOT 72000
#define HID  256
#define NH   4
#define DH   64
#define NL   6
#define BGR  16
#define KTOP 32
#define ETOT 440000

typedef __attribute__((ext_vector_type(16))) __bf16 v16bf;
typedef __attribute__((ext_vector_type(8)))  __bf16 v8bf;
typedef __attribute__((ext_vector_type(8)))  float  v8f;

// ---------------- device helpers ----------------
__device__ __forceinline__ __bf16 f2bf(float f) {
  unsigned u = __float_as_uint(f);
  unsigned r = u + 0x7FFFu + ((u >> 16) & 1u);  // round-to-nearest-even
  unsigned short h = (unsigned short)(r >> 16);
  __bf16 o;
  __builtin_memcpy(&o, &h, 2);
  return o;
}
__device__ __forceinline__ float bf2f(__bf16 h) {
  unsigned short u;
  __builtin_memcpy(&u, &h, 2);
  return __uint_as_float((unsigned)u << 16);
}
__device__ __forceinline__ float gelu_f(float x) {
  float x3 = x * x * x;
  return 0.5f * x * (1.0f + tanhf(0.79788456f * (x + 0.044715f * x3)));
}
__device__ __forceinline__ void atomicMaxF(float* addr, float v) {
  if (v >= 0.0f) atomicMax((int*)addr, __float_as_int(v));
  else           atomicMin((unsigned int*)addr, (unsigned int)__float_as_int(v));
}
#define SHUF16(lo, hi) __builtin_shufflevector(lo, hi, 0,1,2,3,4,5,6,7,8,9,10,11,12,13,14,15)

// ---------------- conversion kernels ----------------
__global__ void k_cvt(const float* __restrict__ s, __bf16* __restrict__ d, int n, int act) {
  int i = blockIdx.x * blockDim.x + threadIdx.x;
  if (i < n) {
    float v = s[i];
    if (act) v = gelu_f(v);
    d[i] = f2bf(v);
  }
}
// Wt[n*K+k] = bf16(W[k*N+n])
__global__ void k_cvt_wT(const float* __restrict__ W, __bf16* __restrict__ Wt, int K, int N) {
  int i = blockIdx.x * blockDim.x + threadIdx.x;
  if (i < K * N) {
    int k = i / N, n = i - k * N;
    Wt[(size_t)n * K + k] = f2bf(W[i]);
  }
}
// relT[h][f][d] = bf16(rel[h][d][f]) for (H,D,D)
__global__ void k_cvt_relT(const float* __restrict__ rel, __bf16* __restrict__ relT) {
  int i = blockIdx.x * blockDim.x + threadIdx.x;  // NH*DH*DH
  if (i < NH * DH * DH) {
    int h = i >> 12, r = i & 4095, d = r >> 6, f = r & 63;
    relT[h * DH * DH + f * DH + d] = f2bf(rel[i]);
  }
}

// ---------------- WMMA bf16 GEMM ----------------
// inputs pre-converted: A bf16 row-major [M,K], Wt bf16 transposed [N,K]
// out_mode: 0 fp32 (optional sigmoid-skip blend with xold), 1 bf16, 2 bf16 + gelu
// wave computes 16x32 (two 16x16 tiles, A fragment reused); block = 4 waves = 128 cols
__global__ void k_gemm(const __bf16* __restrict__ A, const __bf16* __restrict__ Wt,
                       const float* __restrict__ bias, float* __restrict__ C,
                       __bf16* __restrict__ Cb, int M, int K, int N, int out_mode,
                       const float* __restrict__ xold, const float* __restrict__ skip) {
  int wave = threadIdx.x >> 5;
  int lane = threadIdx.x & 31;
  int m0 = blockIdx.x * 16;
  int n0 = blockIdx.y * 128 + wave * 32;
  if (n0 >= N) return;
  int lsub = lane & 15;
  int hi8  = (lane >= 16) ? 8 : 0;
  int row  = m0 + lsub; if (row > M - 1) row = M - 1;   // clamp, keep EXEC uniform
  int col0 = n0 + lsub, col1 = col0 + 16;
  int boff = (lane >= 16) ? 16 : 0;
  const __bf16* arow  = A  + (size_t)row  * K + boff;
  const __bf16* brow0 = Wt + (size_t)col0 * K + boff;
  const __bf16* brow1 = Wt + (size_t)col1 * K + boff;

  v8f acc0 = {}, acc1 = {};
  for (int k0 = 0; k0 < K; k0 += 32) {
    v8bf a0 = *(const v8bf*)(arow + k0);
    v8bf a1 = *(const v8bf*)(arow + k0 + 8);
    v16bf a = SHUF16(a0, a1);
    v8bf b0 = *(const v8bf*)(brow0 + k0);
    v8bf b1 = *(const v8bf*)(brow0 + k0 + 8);
    v16bf b = SHUF16(b0, b1);
    acc0 = __builtin_amdgcn_wmma_f32_16x16x32_bf16(false, a, false, b, (short)0, acc0,
                                                   false, false);
    v8bf c0 = *(const v8bf*)(brow1 + k0);
    v8bf c1 = *(const v8bf*)(brow1 + k0 + 8);
    v16bf bc = SHUF16(c0, c1);
    acc1 = __builtin_amdgcn_wmma_f32_16x16x32_bf16(false, a, false, bc, (short)0, acc1,
                                                   false, false);
  }
  float as = 1.0f, om = 0.0f;
  if (skip) { as = 1.0f / (1.0f + expf(-skip[0])); om = 1.0f - as; }
  v8f accs[2] = {acc0, acc1};
  int cols[2] = {col0, col1};
#pragma unroll
  for (int t = 0; t < 2; ++t) {
    float bv = bias ? bias[cols[t]] : 0.0f;
#pragma unroll
    for (int r = 0; r < 8; ++r) {
      int mrow = m0 + r + hi8;
      if (mrow < M) {
        float o = accs[t][r] + bv;
        if (out_mode == 0) {
          if (skip) o = as * o + om * xold[(size_t)mrow * N + cols[t]];
          C[(size_t)mrow * N + cols[t]] = o;
        } else {
          if (out_mode == 2) o = gelu_f(o);
          Cb[(size_t)mrow * N + cols[t]] = f2bf(o);
        }
      }
    }
  }
}

// ---------------- LayerNorm: fp32 in, bf16 out (wave per row) ----------------
__global__ void k_ln(const float* __restrict__ x, const float* __restrict__ g,
                     const float* __restrict__ b, __bf16* __restrict__ y, int M) {
  int row  = blockIdx.x * (blockDim.x >> 5) + (threadIdx.x >> 5);
  int lane = threadIdx.x & 31;
  if (row >= M) return;
  float v[8];
  float s = 0.0f, s2 = 0.0f;
#pragma unroll
  for (int i = 0; i < 8; ++i) {
    float t = x[(size_t)row * HID + lane + 32 * i];
    v[i] = t; s += t; s2 += t * t;
  }
  for (int m = 16; m >= 1; m >>= 1) { s += __shfl_xor(s, m, 32); s2 += __shfl_xor(s2, m, 32); }
  float mu  = s * (1.0f / 256.0f);
  float var = s2 * (1.0f / 256.0f) - mu * mu;
  float rs  = rsqrtf(var + 1e-5f);
#pragma unroll
  for (int i = 0; i < 8; ++i) {
    int c = lane + 32 * i;
    y[(size_t)row * HID + c] = f2bf((v[i] - mu) * rs * g[c] + b[c]);
  }
}

// ---------------- edge score: kk = gather(kbf)[16,64] @ a_rel[h], score = <q, kk> ----------------
__global__ void k_edge_score(const __bf16* __restrict__ kbf, const float* __restrict__ qbuf,
                             const int* __restrict__ si, const int* __restrict__ di, int E,
                             const __bf16* __restrict__ relT, const float* __restrict__ prel,
                             float* __restrict__ sc, float* __restrict__ segmax,
                             int srcoff, int dstoff, int eoff) {
  int h    = blockIdx.y;
  int lane = threadIdx.x & 31;
  int wid  = threadIdx.x >> 5;
  int e0   = (blockIdx.x * (blockDim.x >> 5) + wid) * 16;
  if (e0 >= E) return;
  int lsub = lane & 15;
  int hi8  = (lane >= 16) ? 8 : 0;
  int boff = (lane >= 16) ? 16 : 0;
  int m    = e0 + lsub; if (m > E - 1) m = E - 1;
  int srow = si[m];
  const __bf16* krow = kbf + (size_t)(srcoff + srow) * HID + h * DH + boff;
  const __bf16* relb = relT + h * DH * DH + boff;

  v8f acc[4] = {};
#pragma unroll
  for (int k0 = 0; k0 < DH; k0 += 32) {
    v8bf a0 = *(const v8bf*)(krow + k0);
    v8bf a1 = *(const v8bf*)(krow + k0 + 8);
    v16bf a = SHUF16(a0, a1);
#pragma unroll
    for (int t = 0; t < 4; ++t) {
      const __bf16* bb = relb + (t * 16 + lsub) * DH + k0;
      v8bf b0 = *(const v8bf*)(bb);
      v8bf b1 = *(const v8bf*)(bb + 8);
      v16bf bm = SHUF16(b0, b1);
      acc[t] = __builtin_amdgcn_wmma_f32_16x16x32_bf16(false, a, false, bm, (short)0,
                                                       acc[t], false, false);
    }
  }
  // dot with gathered q, reduce across the 16 lanes of each half-wave
  float part[8];
#pragma unroll
  for (int r = 0; r < 8; ++r) {
    int me = e0 + r + hi8; if (me > E - 1) me = E - 1;
    const float* qv = qbuf + (size_t)(dstoff + di[me]) * HID + h * DH;
    float p = 0.0f;
#pragma unroll
    for (int t = 0; t < 4; ++t) p += acc[t][r] * qv[t * 16 + lsub];
    part[r] = p;
  }
#pragma unroll
  for (int r = 0; r < 8; ++r)
    for (int msk = 8; msk >= 1; msk >>= 1) part[r] += __shfl_xor(part[r], msk, 32);

  float scale = prel[h] * 0.125f;  // 1/sqrt(64)
#pragma unroll
  for (int r = 0; r < 8; ++r) {
    if (lsub == r) {
      int me = e0 + r + hi8;
      if (me < E) {
        float sv = part[r] * scale;
        sc[(size_t)(eoff + me) * NH + h] = sv;
        atomicMaxF(&segmax[(dstoff + di[me]) * NH + h], sv);
      }
    }
  }
}

// ---------------- exp + segment sum ----------------
__global__ void k_expsum(float* __restrict__ sc, const int* __restrict__ di,
                         const float* __restrict__ segmax, float* __restrict__ segsum,
                         int E, int dstoff, int eoff) {
  int idx = blockIdx.x * blockDim.x + threadIdx.x;
  if (idx >= E * NH) return;
  int e = idx >> 2, h = idx & 3;
  int dg = (dstoff + di[e]) * NH + h;
  float mval = segmax[dg];
  if (mval < -3.0e38f) mval = 0.0f;  // jnp.where(isfinite(m), m, 0)
  float ex = expf(sc[(size_t)(eoff + e) * NH + h] - mval);
  sc[(size_t)(eoff + e) * NH + h] = ex;
  atomicAdd(&segsum[dg], ex);
}

// ---------------- aggregate: vv = gather(vbf) @ m_rel, agg[dst] += alpha * vv ----------------
__global__ void k_edge_aggr(const __bf16* __restrict__ vbf, const int* __restrict__ si,
                            const int* __restrict__ di, int E,
                            const __bf16* __restrict__ relT, const float* __restrict__ sc,
                            const float* __restrict__ segsum, float* __restrict__ agg,
                            int srcoff, int dstoff, int eoff) {
  int h    = blockIdx.y;
  int lane = threadIdx.x & 31;
  int wid  = threadIdx.x >> 5;
  int e0   = (blockIdx.x * (blockDim.x >> 5) + wid) * 16;
  if (e0 >= E) return;
  int lsub = lane & 15;
  int hi8  = (lane >= 16) ? 8 : 0;
  int boff = (lane >= 16) ? 16 : 0;
  int m    = e0 + lsub; if (m > E - 1) m = E - 1;
  int srow = si[m];
  const __bf16* vrow = vbf + (size_t)(srcoff + srow) * HID + h * DH + boff;
  const __bf16* relb = relT + h * DH * DH + boff;

  v8f acc[4] = {};
#pragma unroll
  for (int k0 = 0; k0 < DH; k0 += 32) {
    v8bf a0 = *(const v8bf*)(vrow + k0);
    v8bf a1 = *(const v8bf*)(vrow + k0 + 8);
    v16bf a = SHUF16(a0, a1);
#pragma unroll
    for (int t = 0; t < 4; ++t) {
      const __bf16* bb = relb + (t * 16 + lsub) * DH + k0;
      v8bf b0 = *(const v8bf*)(bb);
      v8bf b1 = *(const v8bf*)(bb + 8);
      v16bf bm = SHUF16(b0, b1);
      acc[t] = __builtin_amdgcn_wmma_f32_16x16x32_bf16(false, a, false, bm, (short)0,
                                                       acc[t], false, false);
    }
  }
#pragma unroll
  for (int r = 0; r < 8; ++r) {
    int me = e0 + r + hi8;
    if (me < E) {
      int drow = dstoff + di[me];
      float w = sc[(size_t)(eoff + me) * NH + h] / (segsum[drow * NH + h] + 1e-16f);
#pragma unroll
      for (int t = 0; t < 4; ++t)
        atomicAdd(&agg[(size_t)drow * HID + h * DH + t * 16 + lsub], acc[t][r] * w);
    }
  }
}

// ---------------- misc elementwise kernels ----------------
__global__ void k_fill(float* p, float v, int n) {
  int i = blockIdx.x * blockDim.x + threadIdx.x;
  if (i < n) p[i] = v;
}
__global__ void k_jkmax(float* __restrict__ xjk, const float* __restrict__ x, int n) {
  int i = blockIdx.x * blockDim.x + threadIdx.x;
  if (i < n) xjk[i] = fmaxf(xjk[i], x[i]);
}
__global__ void k_gemv256(const float* __restrict__ A, const float* __restrict__ w,
                          float* __restrict__ out, int M) {
  int row  = blockIdx.x * (blockDim.x >> 5) + (threadIdx.x >> 5);
  int lane = threadIdx.x & 31;
  if (row >= M) return;
  float s = 0.0f;
  for (int c = lane; c < HID; c += 32) s += A[(size_t)row * HID + c] * w[c];
  for (int m = 16; m >= 1; m >>= 1) s += __shfl_xor(s, m, 32);
  if (lane == 0) out[row] = s;
}

// ---------------- top-k per graph (block per graph) ----------------
__global__ void k_topk(const int* __restrict__ batch, float* __restrict__ scw,
                       int* __restrict__ topidx) {
  int b = blockIdx.x;
  __shared__ float smax[256];
  __shared__ int   sidx[256];
  for (int it = 0; it < KTOP; ++it) {
    float best = -INFINITY; int bi = -1;
    for (int i = threadIdx.x; i < NI_; i += 256) {
      if (batch[i] == b) {
        float s = scw[i];
        if (s > best) { best = s; bi = i; }
      }
    }
    smax[threadIdx.x] = best; sidx[threadIdx.x] = bi;
    __syncthreads();
    for (int st = 128; st > 0; st >>= 1) {
      if (threadIdx.x < st && smax[threadIdx.x + st] > smax[threadIdx.x]) {
        smax[threadIdx.x] = smax[threadIdx.x + st];
        sidx[threadIdx.x] = sidx[threadIdx.x + st];
      }
      __syncthreads();
    }
    if (threadIdx.x == 0) {
      topidx[b * KTOP + it] = sidx[0];
      if (sidx[0] >= 0) scw[sidx[0]] = -INFINITY;
    }
    __syncthreads();
  }
}
__global__ void k_gather_top(const float* __restrict__ xjk, const int* __restrict__ topidx,
                             float* __restrict__ g) {
  int slot = blockIdx.x;  // b*32 + it
  int b = slot >> 5, it = slot & 31;
  int row = topidx[slot];
  int c = threadIdx.x;
  float v = (row >= 0) ? xjk[(size_t)row * HID + c] : 0.0f;
  g[(size_t)b * 8992 + it * HID + c] = v;
}

// ---------------- mean pools ----------------
__global__ void k_poolsum(const float* __restrict__ x, const int* __restrict__ bat,
                          float* __restrict__ psum, float* __restrict__ pcnt,
                          int N, int rowoff, int tt) {
  int idx = blockIdx.x * blockDim.x + threadIdx.x;
  if (idx >= N * HID) return;
  int i = idx / HID, c = idx - i * HID;
  int b = bat[i];
  atomicAdd(&psum[((size_t)b * 3 + tt) * HID + c], x[(size_t)(rowoff + i) * HID + c]);
  if (c == 0) atomicAdd(&pcnt[b * 3 + tt], 1.0f);
}
__global__ void k_poolwrite(const float* __restrict__ psum, const float* __restrict__ pcnt,
                            float* __restrict__ g) {
  int idx = blockIdx.x * blockDim.x + threadIdx.x;
  if (idx >= BGR * 3 * HID) return;
  int c = idx % HID, tt = (idx / HID) % 3, b = idx / (3 * HID);
  float cnt = fmaxf(pcnt[b * 3 + tt], 1.0f);
  g[(size_t)b * 8992 + 8192 + tt * HID + c] = psum[((size_t)b * 3 + tt) * HID + c] / cnt;
}

// ---------------- y-MLP (1->32->32, leaky_relu 0.2) ----------------
__global__ void k_ymlp(const float* __restrict__ yb, const float* __restrict__ W1,
                       const float* __restrict__ b1, const float* __restrict__ W2,
                       const float* __restrict__ b2, float* __restrict__ g) {
  int b = blockIdx.x, j = threadIdx.x;  // 32 threads
  __shared__ float h1[32];
  float t = yb[b] * W1[j] + b1[j];
  h1[j] = (t > 0.0f) ? t : 0.2f * t;
  __syncthreads();
  float s = b2[j];
  for (int k = 0; k < 32; ++k) s += h1[k] * W2[k * 32 + j];
  g[(size_t)b * 8992 + 8960 + j] = s;
}

// ---------------- final 64->1 with gelu on input ----------------
__global__ void k_final(const __bf16* __restrict__ g4, const float* __restrict__ W,
                        const float* __restrict__ b, float* __restrict__ out) {
  int bi = threadIdx.x;
  if (bi < BGR) {
    float s = b[0];
    for (int k = 0; k < 64; ++k) s += gelu_f(bf2f(g4[bi * 64 + k])) * W[k];
    out[bi] = s;
  }
}

// ================= host side =================
static const int ROFF[4]  = {0, 40000, 60000, 68000};
static const int NCNT[4]  = {NI_, NV_, NC_, NB_};
static const int E_CNT[7] = {120000, 80000, 80000, 40000, 40000, 40000, 40000};
static const int E_SRC[7] = {0, 1, 0, 2, 3, 0, 0};
static const int E_DST[7] = {0, 0, 1, 0, 0, 3, 2};
static const int E_OFF[7] = {0, 120000, 200000, 280000, 320000, 360000, 400000};

static inline void gemmB(hipStream_t s, const float* W, __bf16* wt, const __bf16* A,
                         const float* bias, float* C, __bf16* Cb, int M, int K, int N,
                         int out_mode, const float* xold, const float* skip) {
  k_cvt_wT<<<(K * N + 255) / 256, 256, 0, s>>>(W, wt, K, N);
  dim3 g((unsigned)((M + 15) / 16), (unsigned)((N + 127) / 128));
  k_gemm<<<g, 128, 0, s>>>(A, wt, bias, C, Cb, M, K, N, out_mode, xold, skip);
}

extern "C" void kernel_launch(void* const* d_in, const int* in_sizes, int n_in,
                              void* d_out, int out_size, void* d_ws, size_t ws_size,
                              hipStream_t stream) {
  (void)in_sizes; (void)n_in; (void)out_size; (void)ws_size;
  auto F = [&](int i) { return (const float*)d_in[i]; };
  auto I = [&](int i) { return (const int*)d_in[i]; };

  const int P0 = 16;                  // first params leaf
  const int CONV0 = P0 + 16;          // first conv-layer leaf
  const int POOLW = CONV0 + NL * 57;  // 374
  const int YMLP0 = POOLW + 1;        // 375
  const int GMLP0 = YMLP0 + 4;        // 379

  const size_t FEL = (size_t)NTOT * HID;  // 18,432,000
  float* wsf    = (float*)d_ws;
  float* x      = wsf;
  float* qb     = wsf + 1 * FEL;
  float* agg    = wsf + 2 * FEL;
  float* xjk    = wsf + 3 * FEL;
  float* sc     = wsf + 4 * FEL;                // ETOT*NH
  float* segmax = sc + (size_t)ETOT * NH;       // NTOT*NH
  float* segsum = segmax + (size_t)NTOT * NH;
  float* scw    = segsum + (size_t)NTOT * NH;   // NI_
  float* gbuf   = scw + NI_;                    // BGR*8992
  float* g1     = gbuf + (size_t)BGR * 8992;    // 16*512
  float* g2     = g1 + BGR * 512;               // 16*256
  float* g3     = g2 + BGR * 256;               // 16*128
  float* psum   = g3 + BGR * 128;               // 16*3*256
  float* pcnt   = psum + BGR * 3 * HID;         // 16*3
  int*   topidx = (int*)(pcnt + BGR * 3);       // 16*32 ints
  // bf16 region
  __bf16* bfbase = (__bf16*)(topidx + BGR * KTOP);
  __bf16* abf   = bfbase;                        // FEL halves (LN out / gelu(agg) / proj in)
  __bf16* kbf   = abf + FEL;                     // FEL
  __bf16* vbf   = kbf + FEL;                     // FEL
  __bf16* wtbuf = vbf + FEL;                     // up to 8992*512
  __bf16* relTa = wtbuf + (size_t)8992 * 512;    // NH*DH*DH
  __bf16* relTm = relTa + NH * DH * DH;          // NH*DH*DH
  __bf16* gbf   = relTm + NH * DH * DH;          // BGR*8992
  __bf16* g1b   = gbf + (size_t)BGR * 8992;      // 16*512
  __bf16* g2b   = g1b + BGR * 512;               // 16*256
  __bf16* g3b   = g2b + BGR * 256;               // 16*128
  __bf16* g4b   = g3b + BGR * 128;               // 16*64

  // 1) input projections: x[nt] = xd[nt] @ projW + projb
  for (int t = 0; t < 4; ++t) {
    k_cvt<<<(NCNT[t] * 64 + 255) / 256, 256, 0, stream>>>(F(t), abf + (size_t)ROFF[t] * 64,
                                                          NCNT[t] * 64, 0);
    gemmB(stream, F(P0 + t * 2), wtbuf, abf + (size_t)ROFF[t] * 64, F(P0 + t * 2 + 1),
          x + (size_t)ROFF[t] * HID, nullptr, NCNT[t], 64, HID, 0, nullptr, nullptr);
  }

  // 2) init running JK max
  k_fill<<<(int)((FEL + 255) / 256), 256, 0, stream>>>(xjk, -INFINITY, (int)FEL);

  // 3) HGT conv layers
  for (int l = 0; l < NL; ++l) {
    int base = CONV0 + l * 57;
    // LayerNorm per type (shared params), bf16 out into abf
    for (int t = 0; t < 4; ++t)
      k_ln<<<(NCNT[t] + 3) / 4, 128, 0, stream>>>(x + (size_t)ROFF[t] * HID,
                                                  F(P0 + 8 + t * 2), F(P0 + 8 + t * 2 + 1),
                                                  abf + (size_t)ROFF[t] * HID, NCNT[t]);
    // K / Q / V projections (K,V -> bf16 buffers; Q -> fp32)
    for (int t = 0; t < 4; ++t) {
      const __bf16* a = abf + (size_t)ROFF[t] * HID;
      gemmB(stream, F(base + 0 + t * 2), wtbuf, a, F(base + 0 + t * 2 + 1),
            nullptr, kbf + (size_t)ROFF[t] * HID, NCNT[t], HID, HID, 1, nullptr, nullptr);
      gemmB(stream, F(base + 8 + t * 2), wtbuf, a, F(base + 8 + t * 2 + 1),
            qb + (size_t)ROFF[t] * HID, nullptr, NCNT[t], HID, HID, 0, nullptr, nullptr);
      gemmB(stream, F(base + 16 + t * 2), wtbuf, a, F(base + 16 + t * 2 + 1),
            nullptr, vbf + (size_t)ROFF[t] * HID, NCNT[t], HID, HID, 1, nullptr, nullptr);
    }
    // segment buffers + agg
    k_fill<<<(NTOT * NH + 255) / 256, 256, 0, stream>>>(segmax, -INFINITY, NTOT * NH);
    k_fill<<<(NTOT * NH + 255) / 256, 256, 0, stream>>>(segsum, 0.0f, NTOT * NH);
    k_fill<<<(int)((FEL + 255) / 256), 256, 0, stream>>>(agg, 0.0f, (int)FEL);
    // attention scores
    for (int et = 0; et < 7; ++et) {
      const int* ei = I(4 + et);
      int E = E_CNT[et];
      k_cvt_relT<<<(NH * DH * DH + 255) / 256, 256, 0, stream>>>(F(base + 36 + et * 3),
                                                                 relTa);
      dim3 g((unsigned)((E + 63) / 64), NH);
      k_edge_score<<<g, 128, 0, stream>>>(kbf, qb, ei, ei + E, E, relTa,
                                          F(base + 36 + et * 3 + 2), sc, segmax,
                                          ROFF[E_SRC[et]], ROFF[E_DST[et]], E_OFF[et]);
    }
    // softmax exp + segment sums
    for (int et = 0; et < 7; ++et) {
      const int* ei = I(4 + et);
      int E = E_CNT[et];
      k_expsum<<<(E * NH + 255) / 256, 256, 0, stream>>>(sc, ei + E, segmax, segsum, E,
                                                         ROFF[E_DST[et]], E_OFF[et]);
    }
    // weighted aggregation
    for (int et = 0; et < 7; ++et) {
      const int* ei = I(4 + et);
      int E = E_CNT[et];
      k_cvt_relT<<<(NH * DH * DH + 255) / 256, 256, 0, stream>>>(F(base + 36 + et * 3 + 1),
                                                                 relTm);
      dim3 g((unsigned)((E + 63) / 64), NH);
      k_edge_aggr<<<g, 128, 0, stream>>>(vbf, ei, ei + E, E, relTm, sc, segsum, agg,
                                         ROFF[E_SRC[et]], ROFF[E_DST[et]], E_OFF[et]);
    }
    // O projection: x = a*( gelu(agg)@oW + ob ) + (1-a)*x
    k_cvt<<<(int)((FEL + 255) / 256), 256, 0, stream>>>(agg, abf, (int)FEL, 1);
    for (int t = 0; t < 4; ++t)
      gemmB(stream, F(base + 24 + t * 2), wtbuf, abf + (size_t)ROFF[t] * HID,
            F(base + 24 + t * 2 + 1), x + (size_t)ROFF[t] * HID, nullptr, NCNT[t], HID,
            HID, 0, x + (size_t)ROFF[t] * HID, F(base + 32 + t));
    // JK running max
    k_jkmax<<<(int)((FEL + 255) / 256), 256, 0, stream>>>(xjk, x, (int)FEL);
  }

  // 4) pooling
  k_gemv256<<<(NI_ + 3) / 4, 128, 0, stream>>>(xjk, F(POOLW), scw, NI_);
  k_topk<<<BGR, 256, 0, stream>>>(I(11), scw, topidx);
  k_gather_top<<<BGR * KTOP, HID, 0, stream>>>(xjk, topidx, gbuf);
  k_fill<<<(BGR * 3 * HID + BGR * 3 + 255) / 256, 256, 0, stream>>>(psum, 0.0f,
                                                                    BGR * 3 * HID + BGR * 3);
  for (int tt = 0; tt < 3; ++tt) {
    int t = tt + 1;  // var, const, block
    k_poolsum<<<(NCNT[t] * HID + 255) / 256, 256, 0, stream>>>(xjk, I(11 + t), psum, pcnt,
                                                               NCNT[t], ROFF[t], tt);
  }
  k_poolwrite<<<(BGR * 3 * HID + 255) / 256, 256, 0, stream>>>(psum, pcnt, gbuf);
  k_ymlp<<<BGR, 32, 0, stream>>>(F(15), F(YMLP0), F(YMLP0 + 1), F(YMLP0 + 2), F(YMLP0 + 3),
                                 gbuf);

  // 5) graph MLP head (gelu fused into bf16 epilogues)
  k_cvt<<<(BGR * 8992 + 255) / 256, 256, 0, stream>>>(gbuf, gbf, BGR * 8992, 0);
  gemmB(stream, F(GMLP0 + 0), wtbuf, gbf, F(GMLP0 + 1), nullptr, g1b, BGR, 8992, 512, 2,
        nullptr, nullptr);
  gemmB(stream, F(GMLP0 + 2), wtbuf, g1b, F(GMLP0 + 3), nullptr, g2b, BGR, 512, 256, 2,
        nullptr, nullptr);
  gemmB(stream, F(GMLP0 + 4), wtbuf, g2b, F(GMLP0 + 5), nullptr, g3b, BGR, 256, 128, 2,
        nullptr, nullptr);
  gemmB(stream, F(GMLP0 + 6), wtbuf, g3b, F(GMLP0 + 7), nullptr, g4b, BGR, 128, 64, 1,
        nullptr, nullptr);
  k_final<<<1, 64, 0, stream>>>(g4b, F(GMLP0 + 8), F(GMLP0 + 9), (float*)d_out);
  (void)g1; (void)g2; (void)g3;
}
